// GNNRegression_63780264345897
// MI455X (gfx1250) — compile-verified, matrix operand order checked by reading, not run on hardware
//
#include <hip/hip_runtime.h>
#include <hip/hip_bf16.h>

// ---------------------------------------------------------------------------
// GCN (2x GCNConv + global mean pool + linear) for gfx1250 (MI455X).
// GEMMs run on v_wmma_f32_16x16x32_bf16 using a bf16 hi/lo split (3 WMMAs per
// k-step) for near-fp32 accuracy. Edge aggregation uses global_atomic_add_f32
// (h/agg each 102MB -> resident in the 192MB L2).
// ---------------------------------------------------------------------------

#define N_NODES 50000
#define N_EDGES 800000
#define IN_CH   256
#define HID     512
#define N_GRAPHS 64

#define MTILES (N_NODES / 16)   // 3125
#define KT1    (IN_CH / 32)     // 8
#define KT2    (HID / 32)       // 16
#define NT     (HID / 16)       // 32

typedef __attribute__((ext_vector_type(16))) __bf16 v16bf;
typedef __attribute__((ext_vector_type(8)))  float  v8f;

// K-offset of element i (0..15) of a 16-bit A/B WMMA fragment for `lane`.
// (ISA 7.12.2: VGPR v holds K pairs; lanes 16-31 are the +8 K half;
//  VGPRs 4-7 are the +16 K half.)
__device__ __forceinline__ int kmap(int i, int lane) {
    int v = i >> 1, j = i & 1;
    return ((v < 4) ? 0 : 16) + ((lane >= 16) ? 8 : 0) + ((v & 3) << 1) + j;
}

// ---------------------------------------------------------------------------
// Pack fp32 A (row-major MxK) into fragment-major bf16 hi/lo.
// Fragment f = mt*Ktiles + kt; element index = (f*32 + lane)*16 + i.
// ---------------------------------------------------------------------------
__global__ void __launch_bounds__(256)
pack_a_kernel(const float* __restrict__ X, __bf16* __restrict__ Ph,
              __bf16* __restrict__ Pl, int Ktiles, int K, size_t total) {
    size_t t = (size_t)blockIdx.x * blockDim.x + threadIdx.x;
    if (t >= total) return;
    int i    = (int)(t & 15);
    int lane = (int)((t >> 4) & 31);
    size_t f = t >> 9;
    int kt = (int)(f % Ktiles);
    int mt = (int)(f / Ktiles);
    int m = mt * 16 + (lane & 15);
    int k = kt * 32 + kmap(i, lane);
    float x = X[(size_t)m * K + k];
    __bf16 hi = (__bf16)x;
    Ph[t] = hi;
    Pl[t] = (__bf16)(x - (float)hi);
}

// Pack fp32 B = W (row-major KxN) into fragment-major bf16 hi/lo.
// Fragment f = kt*Ntiles + nt.
__global__ void __launch_bounds__(256)
pack_b_kernel(const float* __restrict__ W, __bf16* __restrict__ Ph,
              __bf16* __restrict__ Pl, int Ntiles, int N, size_t total) {
    size_t t = (size_t)blockIdx.x * blockDim.x + threadIdx.x;
    if (t >= total) return;
    int i    = (int)(t & 15);
    int lane = (int)((t >> 4) & 31);
    size_t f = t >> 9;
    int nt = (int)(f % Ntiles);
    int kt = (int)(f / Ntiles);
    int n = nt * 16 + (lane & 15);
    int k = kt * 32 + kmap(i, lane);
    float w = W[(size_t)k * N + n];
    __bf16 hi = (__bf16)w;
    Ph[t] = hi;
    Pl[t] = (__bf16)(w - (float)hi);
}

// ---------------------------------------------------------------------------
// C[M x N] = A[M x K] * B[K x N], bf16x3 split precision on WMMA.
// Block = 256 threads = 8 waves; wave w owns the 16x16 tile (blockIdx.x,
// blockIdx.y*8 + w). A/B are pre-packed fragment-major.
// ---------------------------------------------------------------------------
__global__ void __launch_bounds__(256)
gemm_wmma_bf16x3(const __bf16* __restrict__ Ah, const __bf16* __restrict__ Al,
                 const __bf16* __restrict__ Bh, const __bf16* __restrict__ Bl,
                 float* __restrict__ C, int Ktiles, int Ntiles, int N) {
    const int lane = threadIdx.x & 31;
    const int wave = threadIdx.x >> 5;
    const int mt = blockIdx.x;
    const int nt = blockIdx.y * 8 + wave;
    if (nt >= Ntiles) return;

    const v16bf* aH = (const v16bf*)Ah + (size_t)mt * Ktiles * 32 + lane;
    const v16bf* aL = (const v16bf*)Al + (size_t)mt * Ktiles * 32 + lane;
    const v16bf* bH = (const v16bf*)Bh + (size_t)nt * 32 + lane;
    const v16bf* bL = (const v16bf*)Bl + (size_t)nt * 32 + lane;

    v8f c = {};
    for (int kt = 0; kt < Ktiles; ++kt) {
        v16bf ah = aH[(size_t)kt * 32];
        v16bf al = aL[(size_t)kt * 32];
        v16bf bh = bH[(size_t)kt * Ntiles * 32];
        v16bf bl = bL[(size_t)kt * Ntiles * 32];
        c = __builtin_amdgcn_wmma_f32_16x16x32_bf16(false, ah, false, bh,
                                                    (short)0, c, false, false);
        c = __builtin_amdgcn_wmma_f32_16x16x32_bf16(false, ah, false, bl,
                                                    (short)0, c, false, false);
        c = __builtin_amdgcn_wmma_f32_16x16x32_bf16(false, al, false, bh,
                                                    (short)0, c, false, false);
    }

    // C layout (ISA 7.12.2): row = r + (lane>=16 ? 8 : 0), col = lane & 15.
    const int m0 = mt * 16 + ((lane >= 16) ? 8 : 0);
    const int n  = nt * 16 + (lane & 15);
    #pragma unroll
    for (int r = 0; r < 8; ++r)
        C[(size_t)(m0 + r) * N + n] = c[r];
}

// ---------------------------------------------------------------------------
// Graph plumbing kernels
// ---------------------------------------------------------------------------
__global__ void __launch_bounds__(256)
degree_kernel(const int* __restrict__ dst, const float* __restrict__ ew,
              float* __restrict__ deg, int E) {
    int e = blockIdx.x * blockDim.x + threadIdx.x;
    if (e < E) atomicAdd(&deg[dst[e]], ew[e]);
}

__global__ void __launch_bounds__(256)
dinv_kernel(const float* __restrict__ deg, float* __restrict__ dinv, int n) {
    int i = blockIdx.x * blockDim.x + threadIdx.x;
    if (i < n) dinv[i] = rsqrtf(deg[i] + 1.0f);   // self-loop adds 1; deg >= 1
}

// agg[dst] += dinv[src]*ew*dinv[dst] * h[src, :]   (F = 512, 4 floats/thread)
__global__ void __launch_bounds__(256)
edge_agg_kernel(const int* __restrict__ src, const int* __restrict__ dst,
                const float* __restrict__ ew, const float* __restrict__ dinv,
                const float* __restrict__ h, float* __restrict__ agg, int E) {
    size_t t = (size_t)blockIdx.x * blockDim.x + threadIdx.x;
    size_t total = (size_t)E * (HID / 4);
    if (t >= total) return;
    int e = (int)(t >> 7);            // HID/4 = 128 threads per edge
    int c = (int)(t & 127) << 2;
    int s = src[e], d = dst[e];
    float norm = dinv[s] * ew[e] * dinv[d];
    const float4 v = *(const float4*)(h + (size_t)s * HID + c);
    float* a = agg + (size_t)d * HID + c;
    atomicAdd(a + 0, v.x * norm);
    atomicAdd(a + 1, v.y * norm);
    atomicAdd(a + 2, v.z * norm);
    atomicAdd(a + 3, v.w * norm);
}

// agg = relu(agg + h * dinv^2 + b)   (self-loop + bias + activation, in place)
__global__ void __launch_bounds__(256)
finalize_kernel(float* __restrict__ agg, const float* __restrict__ h,
                const float* __restrict__ dinv, const float* __restrict__ b,
                size_t total) {
    size_t t = (size_t)blockIdx.x * blockDim.x + threadIdx.x;
    if (t >= total) return;
    int f = (int)(t & (HID - 1));
    int i = (int)(t >> 9);
    float dv = dinv[i];
    float v = agg[t] + h[t] * dv * dv + b[f];
    agg[t] = fmaxf(v, 0.0f);
}

// pooled[batch[i], :] += act[i, :]; cnt[batch[i]] += 1
__global__ void __launch_bounds__(256)
pool_kernel(const float* __restrict__ act, const int* __restrict__ batch,
            float* __restrict__ pooled, float* __restrict__ cnt) {
    size_t t = (size_t)blockIdx.x * blockDim.x + threadIdx.x;
    size_t total = (size_t)N_NODES * (HID / 4);
    if (t >= total) return;
    int node = (int)(t >> 7);
    int c = (int)(t & 127) << 2;
    int g = batch[node];
    const float4 v = *(const float4*)(act + (size_t)node * HID + c);
    float* p = pooled + (size_t)g * HID + c;
    atomicAdd(p + 0, v.x);
    atomicAdd(p + 1, v.y);
    atomicAdd(p + 2, v.z);
    atomicAdd(p + 3, v.w);
    if (c == 0) atomicAdd(&cnt[g], 1.0f);
}

// out[g] = (pooled[g,:] . Wl) / max(cnt[g],1) + bl      (OUT_CH == 1)
__global__ void __launch_bounds__(256)
head_kernel(const float* __restrict__ pooled, const float* __restrict__ cnt,
            const float* __restrict__ Wl, const float* __restrict__ bl,
            float* __restrict__ out) {
    __shared__ float red[256];
    int g = blockIdx.x;
    float s = 0.0f;
    for (int f = threadIdx.x; f < HID; f += 256)
        s += pooled[(size_t)g * HID + f] * Wl[f];
    red[threadIdx.x] = s;
    __syncthreads();
    for (int o = 128; o > 0; o >>= 1) {
        if (threadIdx.x < o) red[threadIdx.x] += red[threadIdx.x + o];
        __syncthreads();
    }
    if (threadIdx.x == 0) out[g] = red[0] / fmaxf(cnt[g], 1.0f) + bl[0];
}

// ---------------------------------------------------------------------------
extern "C" void kernel_launch(void* const* d_in, const int* in_sizes, int n_in,
                              void* d_out, int out_size, void* d_ws, size_t ws_size,
                              hipStream_t stream) {
    const float* x     = (const float*)d_in[0];
    const int*   ei    = (const int*)d_in[1];
    const float* ew    = (const float*)d_in[2];
    const int*   batch = (const int*)d_in[3];
    const float* W1    = (const float*)d_in[4];
    const float* b1    = (const float*)d_in[5];
    const float* W2    = (const float*)d_in[6];
    const float* b2    = (const float*)d_in[7];
    const float* Wl    = (const float*)d_in[8];
    const float* bl    = (const float*)d_in[9];
    float* out = (float*)d_out;
    const int* src = ei;
    const int* dst = ei + N_EDGES;

    // Bump-allocate workspace (256B aligned).
    char* cur = (char*)d_ws;
    auto alloc = [&](size_t bytes) -> void* {
        void* p = (void*)cur;
        cur += (bytes + 255) & ~(size_t)255;
        return p;
    };
    float*  h      = (float*) alloc((size_t)N_NODES * HID * 4);   // GEMM out
    float*  agg    = (float*) alloc((size_t)N_NODES * HID * 4);   // agg / act
    float*  deg    = (float*) alloc((size_t)N_NODES * 4);
    float*  dinv   = (float*) alloc((size_t)N_NODES * 4);
    __bf16* pXh    = (__bf16*)alloc((size_t)N_NODES * IN_CH * 2);
    __bf16* pXl    = (__bf16*)alloc((size_t)N_NODES * IN_CH * 2);
    __bf16* pHh    = (__bf16*)alloc((size_t)N_NODES * HID * 2);
    __bf16* pHl    = (__bf16*)alloc((size_t)N_NODES * HID * 2);
    __bf16* pW1h   = (__bf16*)alloc((size_t)IN_CH * HID * 2);
    __bf16* pW1l   = (__bf16*)alloc((size_t)IN_CH * HID * 2);
    __bf16* pW2h   = (__bf16*)alloc((size_t)HID * HID * 2);
    __bf16* pW2l   = (__bf16*)alloc((size_t)HID * HID * 2);
    float*  pooled = (float*) alloc((size_t)N_GRAPHS * HID * 4);
    float*  cnt    = (float*) alloc((size_t)N_GRAPHS * 4);

    const size_t nodeF   = (size_t)N_NODES * HID;                 // 25.6M
    const int    blkNF   = (int)((nodeF + 255) / 256);
    const size_t edgeT   = (size_t)N_EDGES * (HID / 4);
    const int    blkEdge = (int)((edgeT + 255) / 256);

    // ---- degree / normalization -------------------------------------------
    hipMemsetAsync(deg, 0, (size_t)N_NODES * 4, stream);
    degree_kernel<<<(N_EDGES + 255) / 256, 256, 0, stream>>>(dst, ew, deg, N_EDGES);
    dinv_kernel<<<(N_NODES + 255) / 256, 256, 0, stream>>>(deg, dinv, N_NODES);

    // ---- pack weights and X into WMMA fragment order ----------------------
    {
        size_t tW1 = (size_t)IN_CH * HID;
        pack_b_kernel<<<(int)((tW1 + 255) / 256), 256, 0, stream>>>(W1, pW1h, pW1l, NT, HID, tW1);
        size_t tW2 = (size_t)HID * HID;
        pack_b_kernel<<<(int)((tW2 + 255) / 256), 256, 0, stream>>>(W2, pW2h, pW2l, NT, HID, tW2);
        size_t tX = (size_t)N_NODES * IN_CH;
        pack_a_kernel<<<(int)((tX + 255) / 256), 256, 0, stream>>>(x, pXh, pXl, KT1, IN_CH, tX);
    }

    // ---- layer 1: h = X @ W1 ----------------------------------------------
    gemm_wmma_bf16x3<<<dim3(MTILES, NT / 8), 256, 0, stream>>>(
        pXh, pXl, pW1h, pW1l, h, KT1, NT, HID);
    hipMemsetAsync(agg, 0, nodeF * 4, stream);
    edge_agg_kernel<<<blkEdge, 256, 0, stream>>>(src, dst, ew, dinv, h, agg, N_EDGES);
    finalize_kernel<<<blkNF, 256, 0, stream>>>(agg, h, dinv, b1, nodeF);   // agg = act1

    // ---- layer 2: h = act1 @ W2 -------------------------------------------
    pack_a_kernel<<<blkNF, 256, 0, stream>>>(agg, pHh, pHl, KT2, HID, nodeF);
    gemm_wmma_bf16x3<<<dim3(MTILES, NT / 8), 256, 0, stream>>>(
        pHh, pHl, pW2h, pW2l, h, KT2, NT, HID);
    hipMemsetAsync(agg, 0, nodeF * 4, stream);
    edge_agg_kernel<<<blkEdge, 256, 0, stream>>>(src, dst, ew, dinv, h, agg, N_EDGES);
    finalize_kernel<<<blkNF, 256, 0, stream>>>(agg, h, dinv, b2, nodeF);   // agg = act2

    // ---- pool + head -------------------------------------------------------
    hipMemsetAsync(pooled, 0, (size_t)N_GRAPHS * HID * 4, stream);
    hipMemsetAsync(cnt, 0, (size_t)N_GRAPHS * 4, stream);
    pool_kernel<<<(int)(((size_t)N_NODES * (HID / 4) + 255) / 256), 256, 0, stream>>>(
        agg, batch, pooled, cnt);
    head_kernel<<<N_GRAPHS, 256, 0, stream>>>(pooled, cnt, Wl, bl, out);
}
